// FrequencyDomainFusion_42829413875995
// MI455X (gfx1250) — matrix operand order repaired and checked by reading, compiled, not run on hardware
//
#include <hip/hip_runtime.h>
#include <hip/hip_bf16.h>

// ---------------------------------------------------------------------------
// CDNA5 (gfx1250, wave32) implementation of FrequencyDomainFusion.
// GEMM-shaped convs use v_wmma_f32_16x16x32_bf16 (f32 accumulate).
// ---------------------------------------------------------------------------

typedef __attribute__((ext_vector_type(16))) __bf16 v16bf;
typedef __attribute__((ext_vector_type(8)))  float  v8f;

__device__ __forceinline__ __bf16 cvt_bf16(float v) { return (__bf16)v; }

// A-operand k order chosen here (contiguous, not the ISA table order): lanes 0-15 hold K=0..15, lanes 16-31 K=16..31
__device__ __forceinline__ int wmma_kidx_a(int lane, int e) {
  return ((lane >> 4) << 4) + e;
}
// B-operand k order chosen here (contiguous, same as A): lanes 0-15 hold K=0..15, lanes 16-31 K=16..31
__device__ __forceinline__ int wmma_kidx_b(int lane, int e) {
  int j = e >> 1;
  return ((lane >> 4) << 4) + (j << 1) + (e & 1);
}

// ---------------------------------------------------------------------------
// 1x1 conv as GEMM:  Y[b, co, hw] = sum_ci W[co,ci] * X[b,ci,hw] + bias[co]
// Cout == 64 (4 m-tiles), Cin multiple of 32. One wave per 16-pixel tile.
// ---------------------------------------------------------------------------
__global__ void __launch_bounds__(128)
fdf_conv1x1_wmma(const float* __restrict__ X, const float* __restrict__ W,
                 const float* __restrict__ bias, float* __restrict__ Y,
                 int Cin, int Cout, int HW, int B) {
  const int wave = blockIdx.x * (blockDim.x >> 5) + (threadIdx.x >> 5);
  const int lane = threadIdx.x & 31;
  const int tilesPerImg = HW >> 4;
  if (wave >= B * tilesPerImg) return;          // wave-uniform
  const int b    = wave / tilesPerImg;
  const int col0 = (wave - b * tilesPerImg) << 4;
  const int col  = col0 + (lane & 15);
  const float* Xb = X + (size_t)b * Cin * HW;

  v8f acc[4] = {};
  for (int kc = 0; kc < Cin; kc += 32) {
    if (kc + 32 < Cin)
      __builtin_prefetch(&Xb[(size_t)(kc + 32) * HW + col], 0, 1);
    v16bf bfrag;
#pragma unroll
    for (int e = 0; e < 16; ++e) {
      int k = kc + wmma_kidx_b(lane, e);
      bfrag[e] = cvt_bf16(Xb[(size_t)k * HW + col]);
    }
#pragma unroll
    for (int mt = 0; mt < 4; ++mt) {
      int row = (mt << 4) + (lane & 15);
      v16bf afrag;
#pragma unroll
      for (int e = 0; e < 16; ++e) {
        int k = kc + wmma_kidx_a(lane, e);
        afrag[e] = cvt_bf16(W[(size_t)row * Cin + k]);
      }
      acc[mt] = __builtin_amdgcn_wmma_f32_16x16x32_bf16(
          false, afrag, false, bfrag, (short)0, acc[mt], false, false);
    }
  }
#pragma unroll
  for (int mt = 0; mt < 4; ++mt) {
#pragma unroll
    for (int r = 0; r < 8; ++r) {
      int m = (mt << 4) + r + ((lane >> 4) << 3);
      Y[(size_t)b * Cout * HW + (size_t)m * HW + col] = acc[mt][r] + bias[m];
    }
  }
}

// ---------------------------------------------------------------------------
// 3x3 conv (pad=1) as implicit-im2col GEMM: K = Cin*9 (576), M = Cout (9|25).
// Weight layout [Cout, Cin, 3, 3] is exactly row-major [Cout, K].
// ---------------------------------------------------------------------------
template <int MTILES>
__global__ void __launch_bounds__(128)
fdf_conv3x3_wmma(const float* __restrict__ X, const float* __restrict__ W,
                 const float* __restrict__ bias, float* __restrict__ Y,
                 int Cin, int Cout, int H, int Wd, int B) {
  const int HW = H * Wd;
  const int wave = blockIdx.x * (blockDim.x >> 5) + (threadIdx.x >> 5);
  const int lane = threadIdx.x & 31;
  const int tilesPerImg = HW >> 4;
  if (wave >= B * tilesPerImg) return;          // wave-uniform
  const int b    = wave / tilesPerImg;
  const int pix0 = (wave - b * tilesPerImg) << 4;
  const int pix  = pix0 + (lane & 15);
  const int y = pix / Wd, x = pix - (pix / Wd) * Wd;
  const float* Xb = X + (size_t)b * Cin * HW;

  const int K = Cin * 9;
  v8f acc[MTILES] = {};

  for (int kc = 0; kc < K; kc += 32) {
    v16bf bfrag;
#pragma unroll
    for (int e = 0; e < 16; ++e) {
      int k  = kc + wmma_kidx_b(lane, e);
      int ci = k / 9;
      int t  = k - ci * 9;
      int dy = t / 3 - 1;
      int dx = t - (t / 3) * 3 - 1;
      int yy = y + dy, xx = x + dx;
      float v = 0.0f;
      if (yy >= 0 && yy < H && xx >= 0 && xx < Wd)
        v = Xb[(size_t)ci * HW + (size_t)yy * Wd + xx];
      bfrag[e] = cvt_bf16(v);
    }
#pragma unroll
    for (int mt = 0; mt < MTILES; ++mt) {
      int row = (mt << 4) + (lane & 15);
      v16bf afrag;
#pragma unroll
      for (int e = 0; e < 16; ++e) {
        int k = kc + wmma_kidx_a(lane, e);
        float v = (row < Cout) ? W[(size_t)row * K + k] : 0.0f;
        afrag[e] = cvt_bf16(v);
      }
      acc[mt] = __builtin_amdgcn_wmma_f32_16x16x32_bf16(
          false, afrag, false, bfrag, (short)0, acc[mt], false, false);
    }
  }
#pragma unroll
  for (int mt = 0; mt < MTILES; ++mt) {
#pragma unroll
    for (int r = 0; r < 8; ++r) {
      int m = (mt << 4) + r + ((lane >> 4) << 3);
      if (m < Cout)
        Y[(size_t)b * Cout * HW + (size_t)m * HW + pix0 + (lane & 15)] =
            acc[mt][r] + bias[m];
    }
  }
}

// ---------------------------------------------------------------------------
// softmax over k*k taps * Hamming(k)xHamming(k), renormalized.
// Algebraically: out_p = exp(v_p - max) * win_p / sum_q exp(v_q - max)*win_q.
// Optional second input added before normalization (for mask fusion stages).
// ---------------------------------------------------------------------------
__global__ void fdf_normalize(const float* __restrict__ A,
                              const float* __restrict__ Badd,
                              float* __restrict__ Out, int k, int HW, int B) {
  int idx = blockIdx.x * blockDim.x + threadIdx.x;
  if (idx >= B * HW) return;
  int b = idx / HW, p = idx - b * HW;
  const int k2 = k * k;
  const float* Ab = A + (size_t)b * k2 * HW + p;
  const float* Bb = Badd ? Badd + (size_t)b * k2 * HW + p : nullptr;

  float vals[25];
  float mx = -3.4e38f;
  for (int i = 0; i < k2; ++i) {
    float v = Ab[(size_t)i * HW];
    if (Bb) v += Bb[(size_t)i * HW];
    vals[i] = v;
    mx = fmaxf(mx, v);
  }
  const float TWO_PI = 6.28318530717958647692f;
  const float invKm1 = 1.0f / (float)(k - 1);
  float s = 0.0f;
  for (int i = 0; i < k2; ++i) {
    int iy = i / k, ix = i - iy * k;
    float wy = 0.54f - 0.46f * __cosf(TWO_PI * (float)iy * invKm1);
    float wx = 0.54f - 0.46f * __cosf(TWO_PI * (float)ix * invKm1);
    float t  = __expf(vals[i] - mx) * wy * wx;
    vals[i] = t;
    s += t;
  }
  float inv = 1.0f / s;
  float* Ob = Out + (size_t)b * k2 * HW + p;
  for (int i = 0; i < k2; ++i) Ob[(size_t)i * HW] = vals[i] * inv;
}

// ---------------------------------------------------------------------------
// CARAFE k=3, s=1. mode!=0 fuses the residual: out = 2*x - carafe(x, mask).
// One block per (b, c, row); threads across x (coalesced mask + x reads).
// ---------------------------------------------------------------------------
__global__ void fdf_carafe3(const float* __restrict__ X,
                            const float* __restrict__ M,
                            float* __restrict__ Out,
                            int C, int H, int Wd, int B, int mode) {
  int yy = blockIdx.x % H;
  int bc = blockIdx.x / H;
  int c = bc % C, b = bc / C;
  int x = threadIdx.x;
  if (x >= Wd) return;
  size_t HW = (size_t)H * Wd;
  const float* Xc = X + ((size_t)b * C + c) * HW;
  const float* Mb = M + (size_t)b * 9 * HW + (size_t)yy * Wd + x;
  float s = 0.0f;
#pragma unroll
  for (int p = 0; p < 9; ++p) {
    int dy = p / 3 - 1, dx = p % 3 - 1;
    int y2 = yy + dy, x2 = x + dx;
    float xv = (y2 >= 0 && y2 < H && x2 >= 0 && x2 < Wd)
                   ? Xc[(size_t)y2 * Wd + x2] : 0.0f;
    s += xv * Mb[(size_t)p * HW];
  }
  float o = mode ? (2.0f * Xc[(size_t)yy * Wd + x] - s) : s;
  Out[((size_t)b * C + c) * HW + (size_t)yy * Wd + x] = o;
}

// ---------------------------------------------------------------------------
// CARAFE k=5, s=2 upsample: out[b,c,Y,X] = sum_{di,dj} x[b,c,Y/2+di-2,X/2+dj-2]
//                                          * mask[b, di*5+dj, Y, X]
// Mask row (25 x 2*Wl) is staged once in LDS and reused across channels.
// grid = (B*2*Hl, channel-chunks), block = 2*Wl (==128 here).
// ---------------------------------------------------------------------------
__global__ void __launch_bounds__(128)
fdf_carafe5_up(const float* __restrict__ X, const float* __restrict__ M,
               float* __restrict__ Out, int C, int Hl, int Wl, int B,
               int cPerBlock) {
  const int H = Hl * 2, Wd = Wl * 2;
  int Y = blockIdx.x % H;
  int b = blockIdx.x / H;
  int Xo = threadIdx.x;
  if (Xo >= Wd) return;
  size_t HWhi = (size_t)H * Wd;
  size_t HWlo = (size_t)Hl * Wl;

  __shared__ float smask[25 * 128];
  for (int p = 0; p < 25; ++p)
    smask[p * Wd + Xo] =
        M[(size_t)b * 25 * HWhi + (size_t)p * HWhi + (size_t)Y * Wd + Xo];
  __syncthreads();

  int h = Y >> 1, w = Xo >> 1;
  int c0 = blockIdx.y * cPerBlock;
  int c1 = min(C, c0 + cPerBlock);
  for (int c = c0; c < c1; ++c) {
    const float* Xc = X + ((size_t)b * C + c) * HWlo;
    float s = 0.0f;
#pragma unroll
    for (int p = 0; p < 25; ++p) {
      int dy = p / 5 - 2, dx = p % 5 - 2;
      int y2 = h + dy, x2 = w + dx;
      float xv = (y2 >= 0 && y2 < Hl && x2 >= 0 && x2 < Wl)
                     ? Xc[(size_t)y2 * Wl + x2] : 0.0f;
      s += xv * smask[p * Wd + Xo];
    }
    Out[((size_t)b * C + c) * HWhi + (size_t)Y * Wd + Xo] = s;
  }
}

// ---------------------------------------------------------------------------
extern "C" void kernel_launch(void* const* d_in, const int* in_sizes, int n_in,
                              void* d_out, int out_size, void* d_ws,
                              size_t ws_size, hipStream_t stream) {
  constexpr int CHR = 256, CB = 64;
  constexpr int HH = 128, WH = 128, HL = 64, WL = 64;
  constexpr int HWH = HH * WH;   // 16384
  constexpr int HWL = HL * WL;   // 4096
  const int B = in_sizes[0] / (CHR * HWH);

  const float* hr   = (const float*)d_in[0];
  const float* lr   = (const float*)d_in[1];
  const float* hrW  = (const float*)d_in[2];
  const float* hrB  = (const float*)d_in[3];
  const float* lrW  = (const float*)d_in[4];
  const float* lrB  = (const float*)d_in[5];
  const float* lpW  = (const float*)d_in[6];
  const float* lpB  = (const float*)d_in[7];
  const float* hpW  = (const float*)d_in[8];
  const float* hpB  = (const float*)d_in[9];

  float* ws = (float*)d_ws;
  size_t o = 0;
  float* ch          = ws + o; o += (size_t)B * CB * HWH;
  float* cl          = ws + o; o += (size_t)B * CB * HWL;
  float* hf_mask_hr  = ws + o; o += (size_t)B * 9 * HWH;
  float* nhf         = ws + o; o += (size_t)B * 9 * HWH;
  float* hf_enh      = ws + o; o += (size_t)B * CB * HWH;
  float* lf_mask_hr  = ws + o; o += (size_t)B * 25 * HWH;
  float* nlf         = ws + o; o += (size_t)B * 25 * HWH;
  float* lf_mask_lr  = ws + o; o += (size_t)B * 25 * HWL;
  float* up_lf       = ws + o; o += (size_t)B * 25 * HWH;
  float* hf_mask_lr  = ws + o; o += (size_t)B * 9 * HWL;
  float* up_hf       = ws + o; o += (size_t)B * 9 * HWH;
  float* final_hf    = ws + o; o += (size_t)B * 9 * HWH;

  float* out_final_lf = (float*)d_out;                       // [B,25,128,128]
  float* out_enh_hr   = out_final_lf + (size_t)B * 25 * HWH; // [B,256,128,128]
  float* out_up_lr    = out_enh_hr + (size_t)B * CHR * HWH;  // [B,256,128,128]

  const int wavesHR = B * (HWH / 16);     // tiles at 128x128
  const int wavesLR = B * (HWL / 16);     // tiles at 64x64
  const int normHR  = (B * HWH + 255) / 256;

  // 1. ch = 1x1(hr), cl = 1x1(lr)     [WMMA bf16]
  fdf_conv1x1_wmma<<<wavesHR / 4, 128, 0, stream>>>(hr, hrW, hrB, ch, CHR, CB, HWH, B);
  fdf_conv1x1_wmma<<<wavesLR / 4, 128, 0, stream>>>(lr, lrW, lrB, cl, CHR, CB, HWL, B);

  // 2. hf_mask_hr = 3x3(ch, hp) ; nhf = normalize(hf_mask_hr, k=3)
  fdf_conv3x3_wmma<1><<<wavesHR / 4, 128, 0, stream>>>(ch, hpW, hpB, hf_mask_hr, CB, 9, HH, WH, B);
  fdf_normalize<<<normHR, 256, 0, stream>>>(hf_mask_hr, nullptr, nhf, 3, HWH, B);

  // 3. hf_enh = 2*ch - carafe(ch, nhf, 3)
  fdf_carafe3<<<B * CB * HH, WH, 0, stream>>>(ch, nhf, hf_enh, CB, HH, WH, B, 1);

  // 4. lf_mask_hr = 3x3(hf_enh, lp) ; nlf = normalize(., k=5)
  fdf_conv3x3_wmma<2><<<wavesHR / 4, 128, 0, stream>>>(hf_enh, lpW, lpB, lf_mask_hr, CB, 25, HH, WH, B);
  fdf_normalize<<<normHR, 256, 0, stream>>>(lf_mask_hr, nullptr, nlf, 5, HWH, B);

  // 5. lf_mask_lr = 3x3(cl, lp) ; up_lf = carafe(lf_mask_lr, nlf, 5, s=2)
  fdf_conv3x3_wmma<2><<<wavesLR / 4, 128, 0, stream>>>(cl, lpW, lpB, lf_mask_lr, CB, 25, HL, WL, B);
  fdf_carafe5_up<<<dim3(B * HH, 1), 128, 0, stream>>>(lf_mask_lr, nlf, up_lf, 25, HL, WL, B, 25);

  // 6. final_lf = normalize(lf_mask_hr + up_lf, 5)        -> OUTPUT 0
  fdf_normalize<<<normHR, 256, 0, stream>>>(lf_mask_hr, up_lf, out_final_lf, 5, HWH, B);

  // 7. hf_mask_lr = 3x3(cl, hp) ; up_hf = carafe(hf_mask_lr, final_lf, 5, s=2)
  fdf_conv3x3_wmma<1><<<wavesLR / 4, 128, 0, stream>>>(cl, hpW, hpB, hf_mask_lr, CB, 9, HL, WL, B);
  fdf_carafe5_up<<<dim3(B * HH, 1), 128, 0, stream>>>(hf_mask_lr, out_final_lf, up_hf, 9, HL, WL, B, 9);

  // 8. final_hf = normalize(hf_mask_hr + up_hf, 3)
  fdf_normalize<<<normHR, 256, 0, stream>>>(hf_mask_hr, up_hf, final_hf, 3, HWH, B);

  // 9. up_lr = carafe(lr, final_lf, 5, s=2)               -> OUTPUT 2
  fdf_carafe5_up<<<dim3(B * HH, 8), 128, 0, stream>>>(lr, out_final_lf, out_up_lr, CHR, HL, WL, B, 32);

  // 10. enhanced_hr = 2*hr - carafe(hr, final_hf, 3)      -> OUTPUT 1
  fdf_carafe3<<<B * CHR * HH, WH, 0, stream>>>(hr, final_hf, out_enh_hr, CHR, HH, WH, B, 1);
}
